// sglformer_ima2_76776835383807
// MI455X (gfx1250) — compile-verified, matrix operand order checked by reading, not run on hardware
//
#include <hip/hip_runtime.h>
#include <hip/hip_bf16.h>

#define T_ 4
#define B_ 16
#define C_ 256
#define N_ 1024
#define H_ 8
#define D_ 32

typedef __attribute__((ext_vector_type(2)))  float        v2f;
typedef __attribute__((ext_vector_type(8)))  float        v8f;
typedef __attribute__((ext_vector_type(8)))  _Float16     v8h;
typedef __attribute__((ext_vector_type(16))) _Float16     v16h;
typedef __attribute__((ext_vector_type(4)))  unsigned int u32x4;
typedef __attribute__((ext_vector_type(8)))  unsigned int u32x8;

__device__ __forceinline__ v16h pack16(v8h a, v8h b) {
  v16h r;
#pragma unroll
  for (int i = 0; i < 8; ++i) { r[i] = a[i]; r[8 + i] = b[i]; }
  return r;
}

// ---------------------------------------------------------------------------
// TDM load of a 16x256 f32 weight tile (row-major, row stride C_) into LDS
// at offset 0, with 1 DWORD of padding appended per 256-DWORD row so the
// LDS row stride is 257 floats (257 mod 64 = 1 -> conflict-free A-fragment
// reads across lanes).  D# per CDNA5 ISA.
// Issue from one wave only; TENSORcnt is per-wave so that wave waits, then
// the workgroup barrier publishes the LDS tile.
// ---------------------------------------------------------------------------
__device__ __forceinline__ void tdm_load_w_tile(const float* gsrc) {
  unsigned long long ga = (unsigned long long)(uintptr_t)gsrc;
  u32x4 g0;
  g0[0] = 1u;                                            // count=1, user D#
  g0[1] = 0u;                                            // lds_addr = 0
  g0[2] = (unsigned int)ga;                              // global_addr[31:0]
  g0[3] = (unsigned int)((ga >> 32) & 0x01FFFFFFu) | (2u << 30);  // type=2
  u32x8 g1;
  g1[0] = (2u << 16)      // data_size = 4 bytes
        | (1u << 20)      // pad_enable
        | (7u << 22);     // pad_interval code 7 = 256 DWORDs (pad_amount=0 -> 1 DWORD)
  g1[1] = (256u << 16);   // tensor_dim0 = 256  (bits [79:48])
  g1[2] = (16u << 16);    // tensor_dim1 = 16   (bits [111:80])
  g1[3] = (256u << 16);   // tile_dim0 = 256    (bits [127:112])
  g1[4] = 16u;            // tile_dim1 = 16     (bits [143:128])
  g1[5] = 256u;           // tensor_dim0_stride = 256 (bits [207:160])
  g1[6] = 0u;
  g1[7] = 0u;
  asm volatile("tensor_load_to_lds %0, %1" :: "s"(g0), "s"(g1) : "memory");
  __builtin_amdgcn_s_wait_tensorcnt(0);
}

// ---------------------------------------------------------------------------
// K1: Y = W @ X  (f32 WMMA 16x16x4), BN affine, LIF scan over t (state in
// regs).  BRANCH is compile-time: 0=q (writes transposed f16 spikes),
// 1=k, 2=v (row-major f16 spikes; v also writes the vh tuple output).
// ---------------------------------------------------------------------------
template <int BRANCH>
__global__ __launch_bounds__(256)
void sgl_qkv_gemm_bn_lif(const float* __restrict__ x,
                         const float* __restrict__ W,
                         const float* __restrict__ bn_gamma,
                         const float* __restrict__ bn_beta,
                         const float* __restrict__ bn_mean,
                         const float* __restrict__ bn_var,
                         _Float16* __restrict__ qsT,   // [T,B,N,C] (BRANCH 0)
                         _Float16* __restrict__ spk,   // [T,B,C,N] (BRANCH 1/2)
                         float* __restrict__ vh) {     // [T,B,H,N,D] (BRANCH 2)
  __shared__ float Wl[16 * 257];  // padded row-major [m][k], stride 257

  const int lane  = threadIdx.x & 31;
  const int wave  = threadIdx.x >> 5;
  const int nblk  = blockIdx.x;           // N/128
  const int otile = blockIdx.y * 16;      // C/16
  const int b     = blockIdx.z;           // B

  if (wave == 0) tdm_load_w_tile(W + (size_t)otile * C_);
  __syncthreads();

  const int hi   = lane >> 4;   // lane half
  const int lo   = lane & 15;
  const int koff = hi * 2;      // f32 A/B: K pair per lane half
  const int n    = nblk * 128 + wave * 16 + lo;

  float inv[8], add[8];
#pragma unroll
  for (int r = 0; r < 8; ++r) {
    int o = otile + r + 8 * hi;
    float g  = bn_gamma[BRANCH * C_ + o];
    float be = bn_beta [BRANCH * C_ + o];
    float mu = bn_mean [BRANCH * C_ + o];
    float va = bn_var  [BRANCH * C_ + o];
    float iv = g * __frsqrt_rn(va + 1e-5f);
    inv[r] = iv;
    add[r] = be - mu * iv;
  }

  float vmem[8];
#pragma unroll
  for (int r = 0; r < 8; ++r) vmem[r] = 0.f;

  for (int t = 0; t < T_; ++t) {
    const float* xc = x + (size_t)(t * B_ + b) * C_ * N_ + n;
    v8f acc = {};
#pragma unroll 4
    for (int k0 = 0; k0 < C_; k0 += 4) {
      v2f a, bb;
      a[0]  = Wl[lo * 257 + k0 + koff];
      a[1]  = Wl[lo * 257 + k0 + koff + 1];
      bb[0] = xc[(size_t)(k0 + koff) * N_];
      bb[1] = xc[(size_t)(k0 + koff + 1) * N_];
      acc = __builtin_amdgcn_wmma_f32_16x16x4_f32(false, a, false, bb,
                                                  (short)0, acc, false, false);
    }

    if constexpr (BRANCH == 0) {
      v8h qspk;
#pragma unroll
      for (int r = 0; r < 8; ++r) {
        float y  = acc[r] * inv[r] + add[r];
        float vv = vmem[r] + (y - vmem[r]) * 0.5f;   // tau = 2
        float s  = (vv >= 1.0f) ? 1.0f : 0.0f;
        vmem[r]  = vv * (1.0f - s);                  // hard reset
        qspk[r]  = (_Float16)s;
      }
      // 8 consecutive channels per lane -> one 16B store
      _Float16* dst = qsT + ((size_t)(t * B_ + b) * N_ + n) * C_ + otile + 8 * hi;
      *(v8h*)dst = qspk;
    } else {
#pragma unroll
      for (int r = 0; r < 8; ++r) {
        float y  = acc[r] * inv[r] + add[r];
        float vv = vmem[r] + (y - vmem[r]) * 0.5f;
        float s  = (vv >= 1.0f) ? 1.0f : 0.0f;
        vmem[r]  = vv * (1.0f - s);
        int orow = otile + r + 8 * hi;
        spk[((size_t)(t * B_ + b) * C_ + orow) * N_ + n] = (_Float16)s;
        if constexpr (BRANCH == 2) {
          int h = orow >> 5, e = orow & 31;
          vh[(((size_t)(t * B_ + b) * H_ + h) * N_ + n) * D_ + e] = s;
        }
      }
    }
  }
}

// ---------------------------------------------------------------------------
// K2: linear attention per (t,b,h).  Phase 1: kv = k^T v (32x32), exact in
// f16 WMMA (binary spikes -> integer partials <= 1024).  Phase 2: o = q@kv,
// one 16x16x32 WMMA per output tile (K = d = 32).  Writes pre-LIF o * 0.125
// to attn buffer in merged [T,B,C,N] layout.
// ---------------------------------------------------------------------------
__global__ __launch_bounds__(128)
void sgl_linear_attn(const _Float16* __restrict__ qsT,
                     const _Float16* __restrict__ ks,
                     const _Float16* __restrict__ vs,
                     float* __restrict__ attn) {
  __shared__ __align__(16) _Float16 kvT[32 * 40];  // [e][dd], 80B row stride

  const int lane  = threadIdx.x & 31;
  const int wave  = threadIdx.x >> 5;  // 0..3
  const int tb    = blockIdx.x / H_;
  const int h     = blockIdx.x % H_;
  const int hbase = h * D_;
  const int hi = lane >> 4, lo = lane & 15;
  const int koff = hi * 8;  // 16-bit A/B: 8-elt K chunks per lane half

  // ---- phase 1: kv subtile (sm,sn) per wave, contract N=1024 ----
  {
    const int sm = wave >> 1, sn = wave & 1;
    const _Float16* arow = ks + ((size_t)tb * C_ + hbase + sm * 16 + lo) * N_;
    const _Float16* brow = vs + ((size_t)tb * C_ + hbase + sn * 16 + lo) * N_;
    v8f acc = {};
    for (int nc = 0; nc < N_; nc += 32) {
      v16h a  = pack16(*(const v8h*)(arow + nc + koff),
                       *(const v8h*)(arow + nc + 16 + koff));
      v16h bb = pack16(*(const v8h*)(brow + nc + koff),
                       *(const v8h*)(brow + nc + 16 + koff));
      acc = __builtin_amdgcn_wmma_f32_16x16x32_f16(false, a, false, bb,
                                                   (short)0, acc, false, false);
    }
#pragma unroll
    for (int r = 0; r < 8; ++r) {
      int dd = sm * 16 + r + 8 * hi;
      int e  = sn * 16 + lo;
      kvT[e * 40 + dd] = (_Float16)acc[r];  // integers <= 1024: exact in f16
    }
  }
  __syncthreads();

  // ---- phase 2: o[n,e] tiles, 64 x 2 subtiles, 1 WMMA each ----
  for (int it = wave; it < 128; it += 4) {
    int nsub = it >> 1, esub = it & 1;
    const _Float16* arow = qsT + ((size_t)tb * N_ + nsub * 16 + lo) * C_ + hbase;
    v16h a = pack16(*(const v8h*)(arow + koff),
                    *(const v8h*)(arow + 16 + koff));
    const _Float16* brow = &kvT[(esub * 16 + lo) * 40];
    v16h bb = pack16(*(const v8h*)(brow + koff),
                     *(const v8h*)(brow + 16 + koff));
    v8f acc = {};
    acc = __builtin_amdgcn_wmma_f32_16x16x32_f16(false, a, false, bb,
                                                 (short)0, acc, false, false);
#pragma unroll
    for (int r = 0; r < 8; ++r) {
      int nn = nsub * 16 + r + 8 * hi;
      int c  = hbase + esub * 16 + lo;
      attn[((size_t)tb * C_ + c) * N_ + nn] = acc[r] * 0.125f;
    }
  }
}

// ---------------------------------------------------------------------------
// K3: attn LIF (vth = 0.5), in-place scan over t.
// ---------------------------------------------------------------------------
__global__ __launch_bounds__(256)
void sgl_attn_lif(float* __restrict__ buf) {
  const size_t BCN = (size_t)B_ * C_ * N_;
  size_t gid = (size_t)blockIdx.x * 256 + threadIdx.x;
  float v = 0.f;
#pragma unroll
  for (int t = 0; t < T_; ++t) {
    float y = buf[t * BCN + gid];
    v = v + (y - v) * 0.5f;
    float s = (v >= 0.5f) ? 1.f : 0.f;
    v *= (1.f - s);
    buf[t * BCN + gid] = s;
  }
}

// ---------------------------------------------------------------------------
// K4: proj GEMM (f32 WMMA) + bias folded into BN affine + final LIF.
// Output spikes f32 [T,B,C,N] -> d_out (tuple output #1).
// ---------------------------------------------------------------------------
__global__ __launch_bounds__(256)
void sgl_proj_gemm_bn_lif(const float* __restrict__ sin,
                          const float* __restrict__ Wp,
                          const float* __restrict__ bp,
                          const float* __restrict__ bn_gamma,
                          const float* __restrict__ bn_beta,
                          const float* __restrict__ bn_mean,
                          const float* __restrict__ bn_var,
                          float* __restrict__ out) {
  __shared__ float Wl[16 * 257];  // padded row-major [m][k]

  const int lane  = threadIdx.x & 31;
  const int wave  = threadIdx.x >> 5;
  const int nblk  = blockIdx.x;
  const int otile = blockIdx.y * 16;
  const int b     = blockIdx.z;

  if (wave == 0) tdm_load_w_tile(Wp + (size_t)otile * C_);
  __syncthreads();

  const int hi   = lane >> 4;
  const int lo   = lane & 15;
  const int koff = hi * 2;
  const int n    = nblk * 128 + wave * 16 + lo;

  float inv[8], add[8];
#pragma unroll
  for (int r = 0; r < 8; ++r) {
    int o = otile + r + 8 * hi;
    float g  = bn_gamma[3 * C_ + o];
    float be = bn_beta [3 * C_ + o];
    float mu = bn_mean [3 * C_ + o];
    float va = bn_var  [3 * C_ + o];
    float iv = g * __frsqrt_rn(va + 1e-5f);
    inv[r] = iv;
    add[r] = be - mu * iv + bp[o] * iv;  // conv bias folded in
  }

  float vmem[8];
#pragma unroll
  for (int r = 0; r < 8; ++r) vmem[r] = 0.f;

  for (int t = 0; t < T_; ++t) {
    const float* xc = sin + (size_t)(t * B_ + b) * C_ * N_ + n;
    v8f acc = {};
#pragma unroll 4
    for (int k0 = 0; k0 < C_; k0 += 4) {
      v2f a, bb;
      a[0]  = Wl[lo * 257 + k0 + koff];
      a[1]  = Wl[lo * 257 + k0 + koff + 1];
      bb[0] = xc[(size_t)(k0 + koff) * N_];
      bb[1] = xc[(size_t)(k0 + koff + 1) * N_];
      acc = __builtin_amdgcn_wmma_f32_16x16x4_f32(false, a, false, bb,
                                                  (short)0, acc, false, false);
    }
#pragma unroll
    for (int r = 0; r < 8; ++r) {
      float y  = acc[r] * inv[r] + add[r];
      float vv = vmem[r] + (y - vmem[r]) * 0.5f;
      float s  = (vv >= 1.0f) ? 1.0f : 0.0f;
      vmem[r]  = vv * (1.0f - s);
      int orow = otile + r + 8 * hi;
      out[((size_t)(t * B_ + b) * C_ + orow) * N_ + n] = s;
    }
  }
}

// ---------------------------------------------------------------------------
extern "C" void kernel_launch(void* const* d_in, const int* in_sizes, int n_in,
                              void* d_out, int out_size, void* d_ws, size_t ws_size,
                              hipStream_t stream) {
  const float* x  = (const float*)d_in[0];
  // d_in[1] = res_attn (unused by the reference computation)
  const float* Wq = (const float*)d_in[2];
  const float* Wk = (const float*)d_in[3];
  const float* Wv = (const float*)d_in[4];
  const float* Wp = (const float*)d_in[5];
  const float* bp = (const float*)d_in[6];
  const float* g  = (const float*)d_in[7];
  const float* be = (const float*)d_in[8];
  const float* mu = (const float*)d_in[9];
  const float* va = (const float*)d_in[10];

  float* out_o  = (float*)d_out;                              // [T,B,C,N]
  float* out_vh = out_o + (size_t)T_ * B_ * C_ * N_;          // [T,B,H,N,D]

  const size_t SPK = (size_t)T_ * B_ * C_ * N_ * sizeof(_Float16);  // 32 MiB
  char* ws = (char*)d_ws;
  _Float16* qsT  = (_Float16*)ws;                 // [T,B,N,C] f16
  _Float16* ks   = (_Float16*)(ws + SPK);         // [T,B,C,N] f16
  _Float16* vs   = (_Float16*)(ws + 2 * SPK);     // [T,B,C,N] f16
  float*    attn = (float*)   (ws + 3 * SPK);     // [T,B,C,N] f32

  dim3 ggrid(N_ / 128, C_ / 16, B_);

  // K1: fused QKV GEMM + BN + LIF (branch-specialized, no runtime branching)
  sgl_qkv_gemm_bn_lif<0><<<ggrid, 256, 0, stream>>>(x, Wq, g, be, mu, va,
                                                    qsT, ks, out_vh);
  sgl_qkv_gemm_bn_lif<1><<<ggrid, 256, 0, stream>>>(x, Wk, g, be, mu, va,
                                                    qsT, ks, out_vh);
  sgl_qkv_gemm_bn_lif<2><<<ggrid, 256, 0, stream>>>(x, Wv, g, be, mu, va,
                                                    qsT, vs, out_vh);

  // K2: linear attention per (t,b,h)
  sgl_linear_attn<<<dim3(T_ * B_ * H_), 128, 0, stream>>>(qsT, ks, vs, attn);

  // K3: attn LIF (vth=0.5), in place
  sgl_attn_lif<<<dim3((B_ * C_ * N_) / 256), 256, 0, stream>>>(attn);

  // K4: proj GEMM + bias + BN + LIF -> output spikes
  sgl_proj_gemm_bn_lif<<<ggrid, 256, 0, stream>>>(attn, Wp, bp, g, be, mu, va,
                                                  out_o);
}
